// SelfAttentionModule_58841051955611
// MI455X (gfx1250) — compile-verified, matrix operand order checked by reading, not run on hardware
//
#include <hip/hip_runtime.h>
#include <hip/hip_bf16.h>

typedef __attribute__((ext_vector_type(16))) __bf16 v16bf;
typedef __attribute__((ext_vector_type(8)))  __bf16 v8bf;
typedef __attribute__((ext_vector_type(4)))  __bf16 v4bf;
typedef __attribute__((ext_vector_type(2)))  __bf16 v2bf;
typedef __attribute__((ext_vector_type(8)))  float  v8f;
typedef __attribute__((ext_vector_type(4)))  float  v4f;
typedef __attribute__((ext_vector_type(2)))  float  v2f;

#define DEVINL __device__ __forceinline__

// Hardware f32 -> bf16 converts (v_cvt_pk_bf16_f32, RNE).
DEVINL __bf16 f2bf(float f) { return (__bf16)f; }
DEVINL v2bf cvt2(float a, float b) {
  v2f t = {a, b};
  return __builtin_convertvector(t, v2bf);
}
DEVINL v4bf cvt4(v4f v) { return __builtin_convertvector(v, v4bf); }

DEVINL v8f wmma_bf16(v16bf a, v16bf b, v8f c) {
  return __builtin_amdgcn_wmma_f32_16x16x32_bf16(
      /*neg_a=*/false, a, /*neg_b=*/false, b,
      /*c_mod=*/(short)0, c, /*reuse_a=*/false, /*reuse_b=*/false);
}

// Load one 16-element fragment (A or B role) from row-major LDS.
// Element e holds K = base + (e<8 ? 0 : 16) + h*8 + (e&7); contiguous 16B runs.
DEVINL v16bf ld_frag_lds(const __bf16* p, int h) {
  v8bf lo = *(const v8bf*)(p + h * 8);
  v8bf hi = *(const v8bf*)(p + 16 + h * 8);
  v16bf r;
#pragma unroll
  for (int i = 0; i < 8; ++i) { r[i] = lo[i]; r[i + 8] = hi[i]; }
  return r;
}

// A-fragment straight from a global f32 weight row (hardware-converted bf16).
DEVINL v16bf ld_afrag_g32(const float* __restrict__ rowp, int c0) {
  v4f f0 = *(const v4f*)(rowp + c0);
  v4f f1 = *(const v4f*)(rowp + c0 + 4);
  v4f f2 = *(const v4f*)(rowp + c0 + 16);
  v4f f3 = *(const v4f*)(rowp + c0 + 20);
  v4bf b0 = cvt4(f0), b1 = cvt4(f1), b2 = cvt4(f2), b3 = cvt4(f3);
  v16bf r;
#pragma unroll
  for (int i = 0; i < 4; ++i) {
    r[i]      = b0[i];
    r[i + 4]  = b1[i];
    r[i + 8]  = b2[i];
    r[i + 12] = b3[i];
  }
  return r;
}

// LDS layout (dynamic, 296064 B total; HW allows 320KB/workgroup):
//   [0      , 131072) sXT  [n][c] bf16   (X transposed; later per-wave P scratch)
//   [131072 , 262144) sVT  [pix][m] bf16 (V transposed)
//   [262144 , 278528) sQT  [n][32] bf16  (Q, D padded 16->32 with zeros)
//   [278528 , 294912) sKT  [n][32] bf16  (K^T, D padded)
//   [294912 , 295936) sBV  [256] f32
//   [295936 , 296000) sBQ  [16]  f32
//   [296000 , 296064) sBK  [16]  f32
#define SMEM_BYTES 296064

__global__ __launch_bounds__(256)
void SelfAttentionModule_58841051955611_kernel(
    const float* __restrict__ x,  const float* __restrict__ wq,
    const float* __restrict__ bq, const float* __restrict__ wk,
    const float* __restrict__ bk, const float* __restrict__ wv,
    const float* __restrict__ bv, const float* __restrict__ gamma,
    float* __restrict__ out)
{
  extern __shared__ char smem[];
  __bf16* sXT = (__bf16*)(smem);
  __bf16* sVT = (__bf16*)(smem + 131072);
  __bf16* sQT = (__bf16*)(smem + 262144);
  __bf16* sKT = (__bf16*)(smem + 278528);
  float*  sBV = (float*)(smem + 294912);
  float*  sBQ = (float*)(smem + 295936);
  float*  sBK = (float*)(smem + 296000);

  const int tid  = threadIdx.x;
  const int lane = tid & 31;
  const int wave = tid >> 5;
  const int col  = lane & 15;   // fragment column / A-row
  const int h    = lane >> 4;   // lane-half selector for K split
  const int b    = blockIdx.x;

  const float* __restrict__ xb = x + (size_t)b * 65536;
  const v8f vzero = {};

  // Warm caches with the 32 Wv rows (32 KB) this wave consumes in Phase C.
  {
    const char* base = (const char*)(wv + (size_t)(wave * 32) * 256);
#pragma unroll
    for (int i = 0; i < 8; ++i)
      __builtin_prefetch(base + (lane + i * 32) * 128, 0, 0);
  }

  // ---------------- Phase A: stage X (transposed, bf16) + biases ----------
  // 4x4-block transpose: 4 coalesced float4 reads -> 4 packed v4bf stores.
  {
    const v4f* xv = (const v4f*)xb;   // [256 c][64 n4]
#pragma unroll 4
    for (int it = 0; it < 16; ++it) {
      int bidx = tid + it * 256;      // 0..4095 block id
      int cblk = bidx >> 6;           // 0..63
      int nblk = bidx & 63;           // 0..63 (consecutive lanes -> coalesced)
      v4f f0 = xv[(cblk * 4 + 0) * 64 + nblk];
      v4f f1 = xv[(cblk * 4 + 1) * 64 + nblk];
      v4f f2 = xv[(cblk * 4 + 2) * 64 + nblk];
      v4f f3 = xv[(cblk * 4 + 3) * 64 + nblk];
#pragma unroll
      for (int i = 0; i < 4; ++i) {
        v4f t = {f0[i], f1[i], f2[i], f3[i]};
        *(v4bf*)(&sXT[(nblk * 4 + i) * 256 + cblk * 4]) = cvt4(t);
      }
    }
    // zero sQT+sKT (contiguous 32768 B) so D-pad rows 16..31 are zero
    uint4 z = {0u, 0u, 0u, 0u};
    uint4* qz = (uint4*)sQT;
#pragma unroll
    for (int i = 0; i < 8; ++i) qz[tid + i * 256] = z;
    sBV[tid] = bv[tid];
    if (tid < 16) { sBQ[tid] = bq[tid]; sBK[tid] = bk[tid]; }
  }
  __syncthreads();

  // ---------------- Phase B: Q,K projections ([16x256] = W(16x256) * X) ---
  // 32 jobs (16 n-tiles for K, 16 for Q), 4 per wave. Output stored
  // transposed: sQT/sKT[n][d] so scores can use one K=32 WMMA chunk.
#pragma unroll
  for (int jj = 0; jj < 4; ++jj) {
    int job = wave * 4 + jj;
    int isQ = job >> 4;
    int nt  = job & 15;
    const float* __restrict__ wmat = isQ ? wq : wk;
    const float* __restrict__ rowp = wmat + (lane & 15) * 256;  // d = lane&15
    v8f acc = vzero;
#pragma unroll
    for (int ch = 0; ch < 8; ++ch) {
      v16bf a  = ld_afrag_g32(rowp, ch * 32 + h * 8);
      v16bf bb = ld_frag_lds(sXT + (nt * 16 + col) * 256 + ch * 32, h);
      acc = wmma_bf16(a, bb, acc);
    }
    __bf16* dst = isQ ? sQT : sKT;
    const float* sb = isQ ? sBQ : sBK;
    int n = nt * 16 + col;
#pragma unroll
    for (int r = 0; r < 8; r += 2) {
      int d = r + 8 * h;
      *(v2bf*)(&dst[n * 32 + d]) = cvt2(acc[r] + sb[d], acc[r + 1] + sb[d + 1]);
    }
  }
  __syncthreads();

  // ---------------- Phase C: V = Wv * X + bv, stored transposed ----------
  // 16 m-blocks, 2 per wave. Wv streamed from global (shared across blocks,
  // L2-resident); X B-fragments from LDS. Chunk-outer order keeps one
  // A-fragment feeding 16 back-to-back WMMAs.
#pragma unroll
  for (int mm = 0; mm < 2; ++mm) {
    int mb = wave * 2 + mm;
    const float* __restrict__ rowp = wv + (mb * 16 + (lane & 15)) * 256;
    v16bf af[8];
#pragma unroll
    for (int ch = 0; ch < 8; ++ch) af[ch] = ld_afrag_g32(rowp, ch * 32 + h * 8);
    v8f acc[16];
#pragma unroll
    for (int kt = 0; kt < 16; ++kt) acc[kt] = vzero;
#pragma unroll
    for (int ch = 0; ch < 8; ++ch) {
#pragma unroll
      for (int kt = 0; kt < 16; ++kt) {
        v16bf bfrag = ld_frag_lds(sXT + (kt * 16 + col) * 256 + ch * 32, h);
        acc[kt] = wmma_bf16(af[ch], bfrag, acc[kt]);
      }
    }
#pragma unroll
    for (int kt = 0; kt < 16; ++kt) {
      int pix = kt * 16 + col;
#pragma unroll
      for (int r = 0; r < 8; r += 2) {
        int m = mb * 16 + r + 8 * h;
        *(v2bf*)(&sVT[pix * 256 + m]) =
            cvt2(acc[kt][r] + sBV[m], acc[kt][r + 1] + sBV[m + 1]);
      }
    }
  }
  __syncthreads();  // sXT free after this point -> reused as P scratch

  // ---------------- Phase D: scores, softmax, out = P*V, epilogue --------
  const float gm = gamma[0];
  float* __restrict__ ob = out + (size_t)b * 65536;
  __bf16* ps = (__bf16*)(smem + wave * 8192);  // per-wave 16x256 bf16 scratch
#pragma unroll
  for (int bi = 0; bi < 2; ++bi) {
    int rb = wave * 2 + bi;  // 16-row block of scores
    v16bf qf = ld_frag_lds(sQT + (rb * 16 + col) * 32, h);
    v8f s[16];
#pragma unroll
    for (int mt = 0; mt < 16; ++mt) s[mt] = vzero;
#pragma unroll
    for (int mt = 0; mt < 16; ++mt) {
      v16bf kf = ld_frag_lds(sKT + (mt * 16 + col) * 32, h);
      s[mt] = wmma_bf16(qf, kf, s[mt]);
    }
    // Row softmax: element r lives in row rb*16 + r + 8h, cols spread over
    // the 16 lanes of this half; xor masks 1/2/4/8 stay within the half.
#pragma unroll
    for (int r = 0; r < 8; ++r) {
      float mx = s[0][r];
#pragma unroll
      for (int mt = 1; mt < 16; ++mt) mx = fmaxf(mx, s[mt][r]);
      mx = fmaxf(mx, __shfl_xor(mx, 1));
      mx = fmaxf(mx, __shfl_xor(mx, 2));
      mx = fmaxf(mx, __shfl_xor(mx, 4));
      mx = fmaxf(mx, __shfl_xor(mx, 8));
      float sum = 0.f;
#pragma unroll
      for (int mt = 0; mt < 16; ++mt) {
        float p = __expf(s[mt][r] - mx);
        s[mt][r] = p;
        sum += p;
      }
      sum += __shfl_xor(sum, 1);
      sum += __shfl_xor(sum, 2);
      sum += __shfl_xor(sum, 4);
      sum += __shfl_xor(sum, 8);
      float inv = 1.0f / sum;
#pragma unroll
      for (int mt = 0; mt < 16; ++mt) s[mt][r] *= inv;
    }
    // D-layout -> A-layout via per-wave LDS round trip (in-order per wave)
#pragma unroll
    for (int mt = 0; mt < 16; ++mt) {
#pragma unroll
      for (int r = 0; r < 8; ++r)
        ps[(r + 8 * h) * 256 + mt * 16 + col] = f2bf(s[mt][r]);
    }
    // out block = P(16x256) * V(256x256)
    v8f o[16];
#pragma unroll
    for (int kt = 0; kt < 16; ++kt) o[kt] = vzero;
#pragma unroll
    for (int ch = 0; ch < 8; ++ch) {
      v16bf pf = ld_frag_lds(ps + col * 256 + ch * 32, h);
#pragma unroll
      for (int kt = 0; kt < 16; ++kt) {
        v16bf vf = ld_frag_lds(sVT + (kt * 16 + col) * 256 + ch * 32, h);
        o[kt] = wmma_bf16(pf, vf, o[kt]);
      }
    }
    // epilogue: out[b, n, k] = gamma * o + x[b, n, k]  (x re-read; L2-hot).
    // Non-temporal stores: 'out' is write-once, keep x resident in L2.
#pragma unroll
    for (int kt = 0; kt < 16; ++kt) {
#pragma unroll
      for (int r = 0; r < 8; ++r) {
        int idx = (rb * 16 + r + 8 * h) * 256 + kt * 16 + col;
        __builtin_nontemporal_store(gm * o[kt][r] + xb[idx], ob + idx);
      }
    }
  }
}

extern "C" void kernel_launch(void* const* d_in, const int* in_sizes, int n_in,
                              void* d_out, int out_size, void* d_ws, size_t ws_size,
                              hipStream_t stream) {
  const float* x     = (const float*)d_in[0];
  const float* wq    = (const float*)d_in[1];
  const float* bq    = (const float*)d_in[2];
  const float* wk    = (const float*)d_in[3];
  const float* bk    = (const float*)d_in[4];
  const float* wv    = (const float*)d_in[5];
  const float* bv    = (const float*)d_in[6];
  const float* gamma = (const float*)d_in[7];
  float* out = (float*)d_out;
  (void)in_sizes; (void)n_in; (void)out_size; (void)d_ws; (void)ws_size;

  hipLaunchKernelGGL(SelfAttentionModule_58841051955611_kernel,
                     dim3(512), dim3(256), SMEM_BYTES, stream,
                     x, wq, bq, wk, bk, wv, bv, gamma, out);
}
